// SkipConv1dCogVideoXAttnProcessor2_0_58875411694116
// MI455X (gfx1250) — matrix operand order repaired and checked by reading, not compiled, not run on hardware
//
#include <hip/hip_runtime.h>
#include <hip/hip_bf16.h>
#include <math.h>

// ---------------- problem constants ----------------
#define S_TOT  1952
#define TXT_N  224
#define VID_N  1728
#define C_DIM  1920
#define HEADS  30
#define HD_N   64
#define RANK_N 128
#define NF     8
#define HH_N   12
#define WW_N   18
#define NP_N   (HH_N * WW_N)   // 216

typedef __attribute__((ext_vector_type(16))) __bf16 bf16x16;
typedef __attribute__((ext_vector_type(8)))  __bf16 bf16x8;
typedef __attribute__((ext_vector_type(8)))  float  f32x8;

// ---------------- device helpers ----------------
__device__ __forceinline__ f32x8 zero8() {
  f32x8 z;
#pragma unroll
  for (int e = 0; e < 8; ++e) z[e] = 0.0f;
  return z;
}

__device__ __forceinline__ f32x8 wmma_bf16(bf16x16 a, bf16x16 b, f32x8 c) {
  // D = A(16x32 bf16) * B(32x16 bf16) + C(16x16 f32)
  return __builtin_amdgcn_wmma_f32_16x16x32_bf16(false, a, false, b,
                                                 (short)0, c, false, false);
}

// A-operand fragment (16-bit A 16x32; this file's k order, not the ISA table order):
// lane L: row = L%16, half = L/16; elements 0..15 = K[koff+half*16 .. +15]
// (contiguous, matches B operand)  -> two ds/global b128 loads.
__device__ __forceinline__ bf16x16 make_a_frag(const __bf16* rowptr, int koff, int hw) {
  bf16x8 lo = *(const bf16x8*)(rowptr + koff + hw * 16);
  bf16x8 hi = *(const bf16x8*)(rowptr + koff + hw * 16 + 8);
  bf16x16 f;
#pragma unroll
  for (int i = 0; i < 8; ++i) { f[i] = lo[i]; f[i + 8] = hi[i]; }
  return f;
}

// wave-local LDS ordering: CDNA5 split counter
__device__ __forceinline__ void wave_lds_fence() {
  asm volatile("s_wait_dscnt 0" ::: "memory");
}

// ---------------- scalar prep kernels ----------------
__global__ void cvt_bf16_kernel(const float* __restrict__ in,
                                __bf16* __restrict__ out, int n) {
  int i = blockIdx.x * blockDim.x + threadIdx.x;
  if (i < n) out[i] = (__bf16)in[i];
}

__global__ void enc_copy_kernel(const float* __restrict__ enc,
                                __bf16* __restrict__ hs_b) {
  int i = blockIdx.x * blockDim.x + threadIdx.x;
  if (i < TXT_N * C_DIM) hs_b[i] = (__bf16)enc[i];
}

// conv1 (C=1920 -> RANK=128, ksize 3, pad 1, over frame axis) + exact GELU
__global__ void conv1_gelu_kernel(const float* __restrict__ hidden,
                                  const float* __restrict__ w1,
                                  float* __restrict__ h1) {
  int p = blockIdx.x;        // 0..215 spatial
  int r = threadIdx.x;       // 0..127 rank
  int hh = p / WW_N, ww = p % WW_N;
  float acc[NF];
#pragma unroll
  for (int f = 0; f < NF; ++f) acc[f] = 0.0f;
  const float* wrow = w1 + (size_t)r * C_DIM * 3;
  for (int c = 0; c < C_DIM; ++c) {
    float xc[NF];
#pragma unroll
    for (int f = 0; f < NF; ++f)
      xc[f] = hidden[((size_t)(f * HH_N + hh) * WW_N + ww) * C_DIM + c];
    float wa = wrow[c * 3 + 0], wb = wrow[c * 3 + 1], wc = wrow[c * 3 + 2];
#pragma unroll
    for (int f = 0; f < NF; ++f) {
      float s = wb * xc[f];
      if (f > 0)      s += wa * xc[f - 1];
      if (f < NF - 1) s += wc * xc[f + 1];
      acc[f] += s;
    }
  }
#pragma unroll
  for (int f = 0; f < NF; ++f) {
    float v = acc[f];
    v = 0.5f * v * (1.0f + erff(v * 0.70710678118654752f));  // exact GELU
    h1[((size_t)p * RANK_N + r) * NF + f] = v;
  }
}

// conv2 (RANK=128 -> C=1920) + residual add; writes video rows of hs (bf16)
__global__ void conv2_add_kernel(const float* __restrict__ hidden,
                                 const float* __restrict__ h1,
                                 const float* __restrict__ w2,
                                 __bf16* __restrict__ hs_b) {
  int id = blockIdx.x * blockDim.x + threadIdx.x;
  if (id >= NP_N * C_DIM) return;
  int c = id % C_DIM, p = id / C_DIM;
  int hh = p / WW_N, ww = p % WW_N;
  float acc[NF];
#pragma unroll
  for (int f = 0; f < NF; ++f) acc[f] = 0.0f;
  const float* wrow = w2 + (size_t)c * RANK_N * 3;
  const float* hp = h1 + (size_t)p * RANK_N * NF;
  for (int r = 0; r < RANK_N; ++r) {
    float hr[NF];
#pragma unroll
    for (int f = 0; f < NF; ++f) hr[f] = hp[r * NF + f];
    float wa = wrow[r * 3 + 0], wb = wrow[r * 3 + 1], wc = wrow[r * 3 + 2];
#pragma unroll
    for (int f = 0; f < NF; ++f) {
      float s = wb * hr[f];
      if (f > 0)      s += wa * hr[f - 1];
      if (f < NF - 1) s += wc * hr[f + 1];
      acc[f] += s;
    }
  }
#pragma unroll
  for (int f = 0; f < NF; ++f) {
    int vid = (f * HH_N + hh) * WW_N + ww;
    float val = hidden[(size_t)vid * C_DIM + c] + acc[f];
    hs_b[(size_t)(TXT_N + vid) * C_DIM + c] = (__bf16)val;
  }
}

// interleaved RoPE, in place on the video rows of q and k (bf16)
__global__ void rope_kernel(__bf16* __restrict__ qb, __bf16* __restrict__ kb,
                            const float* __restrict__ cosp,
                            const float* __restrict__ sinp) {
  int i = blockIdx.x * blockDim.x + threadIdx.x;
  const int total = VID_N * HEADS * (HD_N / 2);
  if (i >= total) return;
  int d  = i % (HD_N / 2);
  int t  = i / (HD_N / 2);
  int hh = t % HEADS;
  int s  = t / HEADS;
  float c0 = cosp[s * HD_N + 2 * d],     s0 = sinp[s * HD_N + 2 * d];
  float c1 = cosp[s * HD_N + 2 * d + 1], s1 = sinp[s * HD_N + 2 * d + 1];
  size_t off = (size_t)(TXT_N + s) * C_DIM + hh * HD_N + 2 * d;
  float q0 = (float)qb[off], q1 = (float)qb[off + 1];
  qb[off]     = (__bf16)(q0 * c0 - q1 * s0);
  qb[off + 1] = (__bf16)(q1 * c1 + q0 * s1);
  float k0 = (float)kb[off], k1 = (float)kb[off + 1];
  kb[off]     = (__bf16)(k0 * c0 - k1 * s0);
  kb[off + 1] = (__bf16)(k1 * c1 + k0 * s1);
}

// ---------------- WMMA GEMM: Out[M,N] = A[M,K] @ W[N,K]^T + bias ----------------
// mode 0: bf16 row-major out; mode 1: f32 to d_out with (video,text) row perm;
// mode 2: bf16 transposed out (out[col*M + row]) for V.
__global__ __launch_bounds__(256) void gemm_wmma_kernel(
    const __bf16* __restrict__ A, const __bf16* __restrict__ W,
    const float* __restrict__ bias,
    __bf16* __restrict__ outb, float* __restrict__ outf,
    int M, int N, int K, int mode) {
  const int lane = threadIdx.x & 31;
  const int hw   = lane >> 4;
  const int l16  = lane & 15;
  const int wave = threadIdx.x >> 5;
  const int mtiles = M >> 5;   // tiles of 32 rows
  const int ntiles = N >> 6;   // tiles of 64 cols
  int tile = blockIdx.x * 8 + wave;
  if (tile >= mtiles * ntiles) return;  // wave-uniform guard
  int mt = tile % mtiles, nt = tile / mtiles;
  int m0 = mt * 32, n0 = nt * 64;

  f32x8 acc[2][4];
#pragma unroll
  for (int i = 0; i < 2; ++i)
#pragma unroll
    for (int j = 0; j < 4; ++j) acc[i][j] = zero8();

  const __bf16* arow0 = A + (size_t)(m0 + l16) * K;
  const __bf16* arow1 = A + (size_t)(m0 + 16 + l16) * K;

  for (int k = 0; k < K; k += 32) {
    if (k + 32 < K) {
      __builtin_prefetch(arow0 + k + 32, 0, 1);   // global_prefetch_b8
      __builtin_prefetch(arow1 + k + 32, 0, 1);
    }
    bf16x16 a0 = make_a_frag(arow0, k, hw);
    bf16x16 a1 = make_a_frag(arow1, k, hw);
#pragma unroll
    for (int j = 0; j < 4; ++j) {
      // B-operand: lane holds col n = n0+16j+l16; 16 contiguous K at k+16*hw
      const __bf16* wp = W + (size_t)(n0 + 16 * j + l16) * K + k + 16 * hw;
      bf16x16 b = *(const bf16x16*)wp;
      acc[0][j] = wmma_bf16(a0, b, acc[0][j]);
      acc[1][j] = wmma_bf16(a1, b, acc[1][j]);
    }
  }

#pragma unroll
  for (int j = 0; j < 4; ++j) {
    int col = n0 + 16 * j + l16;
    float bval = bias[col];
#pragma unroll
    for (int i = 0; i < 2; ++i)
#pragma unroll
      for (int r = 0; r < 8; ++r) {
        int row = m0 + 16 * i + r + 8 * hw;   // C/D layout: M = r + 8*half
        float v = acc[i][j][r] + bval;
        if (mode == 0) {
          outb[(size_t)row * N + col] = (__bf16)v;
        } else if (mode == 2) {
          outb[(size_t)col * M + row] = (__bf16)v;
        } else {
          int prow = (row < TXT_N) ? row + VID_N : row - TXT_N;
          outf[(size_t)prow * N + col] = v;
        }
      }
  }
}

// ---------------- flash attention (wave = 16 query rows of one head) --------
__global__ __launch_bounds__(256) void flash_attn_kernel(
    const __bf16* __restrict__ qb, const __bf16* __restrict__ kb,
    const __bf16* __restrict__ vT, __bf16* __restrict__ ob) {
  __shared__ __align__(32) float  sS[8][16][32];
  __shared__ __align__(32) __bf16 sP[8][16][32];
  __shared__ float sAlpha[8][16];
  __shared__ float sL[8][16];

  const int lane = threadIdx.x & 31;
  const int hw   = lane >> 4;
  const int l16  = lane & 15;
  const int wave = threadIdx.x >> 5;

  const int qTiles = S_TOT / 16;           // 122
  int id = blockIdx.x * 8 + wave;
  if (id >= HEADS * qTiles) return;        // wave-uniform, no block barriers used
  int h  = id / qTiles;
  int qt = id % qTiles;

  // Q fragments for this wave's 16 rows (K = HD = 64 -> two 32-wide slices)
  const __bf16* qrow = qb + (size_t)(qt * 16 + l16) * C_DIM + h * HD_N;
  bf16x16 aq0 = make_a_frag(qrow, 0, hw);
  bf16x16 aq1 = make_a_frag(qrow, 32, hw);

  f32x8 oacc[4];
#pragma unroll
  for (int j = 0; j < 4; ++j) oacc[j] = zero8();

  float m = -3.0e38f, l = 0.0f;
  const float scale = 0.125f;              // HD^-0.5

  for (int ch = 0; ch < S_TOT / 32; ++ch) {
    int key0 = ch * 32;
    // ---- scores: two 16x16 tiles, K=64 each (2 WMMAs) ----
#pragma unroll
    for (int kt = 0; kt < 2; ++kt) {
      const __bf16* kp =
          kb + (size_t)(key0 + kt * 16 + l16) * C_DIM + h * HD_N + 16 * hw;
      bf16x16 bk0 = *(const bf16x16*)(kp);
      bf16x16 bk1 = *(const bf16x16*)(kp + 32);
      f32x8 s = zero8();
      s = wmma_bf16(aq0, bk0, s);
      s = wmma_bf16(aq1, bk1, s);
#pragma unroll
      for (int r = 0; r < 8; ++r)
        sS[wave][r + 8 * hw][kt * 16 + l16] = s[r] * scale;   // D -> row-major
    }
    wave_lds_fence();

    // ---- online softmax, lane-per-row ----
    if (lane < 16) {
      float sv[32];
      float tmax = -3.0e38f;
#pragma unroll
      for (int j = 0; j < 32; ++j) {
        sv[j] = sS[wave][lane][j];
        tmax = fmaxf(tmax, sv[j]);
      }
      float mnew  = fmaxf(m, tmax);
      float alpha = __expf(m - mnew);      // first chunk: exp(-inf)=0
      float rsum  = 0.0f;
#pragma unroll
      for (int j = 0; j < 32; ++j) {
        float pv = __expf(sv[j] - mnew);
        rsum += pv;
        sP[wave][lane][j] = (__bf16)pv;
      }
      l = l * alpha + rsum;
      m = mnew;
      sAlpha[wave][lane] = alpha;
    }
    wave_lds_fence();

    // ---- rescale accumulator ----
#pragma unroll
    for (int r = 0; r < 8; ++r) {
      float al = sAlpha[wave][r + 8 * hw];
#pragma unroll
      for (int j = 0; j < 4; ++j) oacc[j][r] *= al;
    }

    // ---- P @ V : P(16x32) from LDS in A-layout, V from transposed buffer ----
    bf16x16 ap = make_a_frag(&sP[wave][l16][0], 0, hw);
#pragma unroll
    for (int j = 0; j < 4; ++j) {
      const __bf16* vp =
          vT + (size_t)(h * HD_N + 16 * j + l16) * S_TOT + key0 + 16 * hw;
      bf16x16 bv = *(const bf16x16*)vp;
      oacc[j] = wmma_bf16(ap, bv, oacc[j]);
    }
  }

  if (lane < 16) sL[wave][lane] = 1.0f / l;
  wave_lds_fence();
#pragma unroll
  for (int r = 0; r < 8; ++r) {
    float linv = sL[wave][r + 8 * hw];
    int row = qt * 16 + r + 8 * hw;
#pragma unroll
    for (int j = 0; j < 4; ++j) {
      int col = h * HD_N + 16 * j + l16;
      ob[(size_t)row * C_DIM + col] = (__bf16)(oacc[j][r] * linv);
    }
  }
}

// ---------------- host launcher ----------------
extern "C" void kernel_launch(void* const* d_in, const int* in_sizes, int n_in,
                              void* d_out, int out_size, void* d_ws, size_t ws_size,
                              hipStream_t stream) {
  (void)in_sizes; (void)n_in; (void)out_size; (void)ws_size;
  const float* hidden = (const float*)d_in[0];
  const float* enc    = (const float*)d_in[1];
  const float* cosp   = (const float*)d_in[2];
  const float* sinp   = (const float*)d_in[3];
  const float* w1     = (const float*)d_in[4];
  const float* w2     = (const float*)d_in[5];
  const float* Wq     = (const float*)d_in[6];
  const float* bq     = (const float*)d_in[7];
  const float* Wk     = (const float*)d_in[8];
  const float* bk     = (const float*)d_in[9];
  const float* Wv     = (const float*)d_in[10];
  const float* bv     = (const float*)d_in[11];
  const float* Wo     = (const float*)d_in[12];
  const float* bo     = (const float*)d_in[13];

  char* ws = (char*)d_ws;
  size_t off = 0;
  auto carve = [&](size_t bytes) -> void* {
    void* p = ws + off;
    off = (off + bytes + 255) & ~(size_t)255;
    return p;
  };
  const size_t SC2 = (size_t)S_TOT * C_DIM * sizeof(__bf16);
  const size_t CC2 = (size_t)C_DIM * C_DIM * sizeof(__bf16);
  __bf16* hs_b = (__bf16*)carve(SC2);
  __bf16* wq_b = (__bf16*)carve(CC2);
  __bf16* wk_b = (__bf16*)carve(CC2);
  __bf16* wv_b = (__bf16*)carve(CC2);
  __bf16* wo_b = (__bf16*)carve(CC2);
  __bf16* qbuf = (__bf16*)carve(SC2);
  __bf16* kbuf = (__bf16*)carve(SC2);
  __bf16* vTbf = (__bf16*)carve(SC2);      // [C][S] transposed
  __bf16* obuf = (__bf16*)carve(SC2);
  float*  h1   = (float*)carve((size_t)NP_N * RANK_N * NF * sizeof(float));

  // 1) weights -> bf16
  const int nW = C_DIM * C_DIM;
  cvt_bf16_kernel<<<(nW + 255) / 256, 256, 0, stream>>>(Wq, wq_b, nW);
  cvt_bf16_kernel<<<(nW + 255) / 256, 256, 0, stream>>>(Wk, wk_b, nW);
  cvt_bf16_kernel<<<(nW + 255) / 256, 256, 0, stream>>>(Wv, wv_b, nW);
  cvt_bf16_kernel<<<(nW + 255) / 256, 256, 0, stream>>>(Wo, wo_b, nW);

  // 2) skip branch + build concatenated hs (bf16)
  conv1_gelu_kernel<<<NP_N, RANK_N, 0, stream>>>(hidden, w1, h1);
  conv2_add_kernel<<<(NP_N * C_DIM + 255) / 256, 256, 0, stream>>>(hidden, h1, w2, hs_b);
  enc_copy_kernel<<<(TXT_N * C_DIM + 255) / 256, 256, 0, stream>>>(enc, hs_b);

  // 3) Q/K/V projections (WMMA). V stored transposed for the P@V B-operand.
  const int gemmTiles  = (S_TOT / 32) * (C_DIM / 64);         // 1830
  const int gemmBlocks = (gemmTiles + 7) / 8;                 // 229
  gemm_wmma_kernel<<<gemmBlocks, 256, 0, stream>>>(hs_b, wq_b, bq, qbuf, nullptr,
                                                   S_TOT, C_DIM, C_DIM, 0);
  gemm_wmma_kernel<<<gemmBlocks, 256, 0, stream>>>(hs_b, wk_b, bk, kbuf, nullptr,
                                                   S_TOT, C_DIM, C_DIM, 0);
  gemm_wmma_kernel<<<gemmBlocks, 256, 0, stream>>>(hs_b, wv_b, bv, vTbf, nullptr,
                                                   S_TOT, C_DIM, C_DIM, 2);

  // 4) RoPE on video rows of q, k
  const int nRope = VID_N * HEADS * (HD_N / 2);
  rope_kernel<<<(nRope + 255) / 256, 256, 0, stream>>>(qbuf, kbuf, cosp, sinp);

  // 5) flash attention
  const int attnTiles  = HEADS * (S_TOT / 16);                // 3660
  const int attnBlocks = (attnTiles + 7) / 8;                 // 458
  flash_attn_kernel<<<attnBlocks, 256, 0, stream>>>(qbuf, kbuf, vTbf, obuf);

  // 6) output projection -> f32 d_out with (video, text) row permutation
  gemm_wmma_kernel<<<gemmBlocks, 256, 0, stream>>>(obuf, wo_b, bo, nullptr,
                                                   (float*)d_out,
                                                   S_TOT, C_DIM, C_DIM, 1);
}
